// GCNGraphClassification_4733053960245
// MI455X (gfx1250) — matrix operand order repaired and checked by reading, not compiled, not run on hardware
//
#include <hip/hip_runtime.h>
#include <hip/hip_bf16.h>
#include <cstddef>

typedef __bf16 bf16;
typedef __attribute__((ext_vector_type(16))) __bf16 v16bf;
typedef __attribute__((ext_vector_type(8)))  __bf16 v8bf;
typedef __attribute__((ext_vector_type(8)))  float  v8f;

#define MT 5   // M tiles (16 rows each) per wave; 50000 rows = 625 * 80

// ---------------- degree + norm ----------------
__global__ void k_degree(const int* __restrict__ src, const int* __restrict__ dst,
                         float* __restrict__ degs, float* __restrict__ degd, int E) {
    int e = blockIdx.x * blockDim.x + threadIdx.x;
    if (e < E) {
        unsafeAtomicAdd(&degs[src[e]], 1.0f);
        unsafeAtomicAdd(&degd[dst[e]], 1.0f);
    }
}

__global__ void k_rsqrt_clamp(float* __restrict__ d, int n) {
    int i = blockIdx.x * blockDim.x + threadIdx.x;
    if (i < n) d[i] = rsqrtf(fmaxf(d[i], 1.0f));
}

// ---------------- weight cast + pack into WMMA B-fragment order ----------------
// Bpack[((nT*(K/32)+kt)*32 + lane)*16 + e] = (bf16)W[(kt*32 + (lane>>4)*16 + e)*N + nT*16 + (lane&15)]
// => per-lane fragment is 16 contiguous bf16 (32B) = one aligned v16bf load.
__global__ void k_packW(const float* __restrict__ W, bf16* __restrict__ Bp,
                        int K, int N) {
    int i = blockIdx.x * blockDim.x + threadIdx.x;
    if (i >= K * N) return;
    int e    = i & 15;
    int lane = (i >> 4) & 31;
    int frag = i >> 9;                  // nT*(K/32) + kt
    int ksteps = K >> 5;
    int nT = frag / ksteps;
    int kt = frag - nT * ksteps;
    int col  = nT * 16 + (lane & 15);
    int krow = kt * 32 + ((lane >> 4) << 4) + e;
    Bp[i] = (bf16)W[(size_t)krow * N + col];
}

// ---------------- prep: xs = bf16(x * cj) ----------------
__global__ void k_prep1(const float* __restrict__ x, const float* __restrict__ cj,
                        bf16* __restrict__ xs, int total, int F) {
    int i = blockIdx.x * blockDim.x + threadIdx.x;
    if (i >= total) return;
    int n = i / F;
    xs[i] = (bf16)(x[i] * cj[n]);
}

// ---------------- bf16 WMMA GEMM: C[M,N] = A[M,K] @ B[K,N] (B pre-packed) ----
// wave -> 5 stacked 16x16 C tiles (80 rows x 16 cols); 8 waves/block -> 128 cols
// grid = (M/80, N/128)
__global__ __launch_bounds__(256)
void k_gemm_bf16(const bf16* __restrict__ A, const bf16* __restrict__ Bp,
                 float* __restrict__ C, int M, int N, int K) {
    const int lane = threadIdx.x & 31;
    const int wave = threadIdx.x >> 5;
    const int m0 = blockIdx.x * (16 * MT);
    const int nT = blockIdx.y * 8 + wave;
    const int n0 = nT * 16;
    if (n0 + 16 > N || m0 + 16 * MT > M) return;

    const int row = lane & 15;           // A row / C col within tile
    const int kA  = (lane >> 4) << 4;    // A: lanes16-31 start at K+16
    const int ksteps = K >> 5;

    v8f acc[MT] = {};
    const bf16* Bbase = Bp + ((size_t)nT * ksteps * 32 + lane) * 16;

    for (int kt = 0; kt < ksteps; ++kt) {
        v16bf b = *(const v16bf*)(Bbase + (size_t)kt * 512);   // 32B contiguous
        const int k0 = kt * 32;
#pragma unroll
        for (int t = 0; t < MT; ++t) {
            const bf16* Arow = A + (size_t)(m0 + t * 16 + row) * K + k0 + kA;
            v8bf alo = *(const v8bf*)(Arow);
            v8bf ahi = *(const v8bf*)(Arow + 8);
            v16bf a;
#pragma unroll
            for (int j = 0; j < 8; ++j) { a[j] = alo[j]; a[8 + j] = ahi[j]; }
            acc[t] = __builtin_amdgcn_wmma_f32_16x16x32_bf16(
                false, a, false, b, (short)0, acc[t], false, false);
        }
    }

    const int hi8 = (lane >> 4) << 3;
#pragma unroll
    for (int t = 0; t < MT; ++t) {
#pragma unroll
        for (int v = 0; v < 8; ++v)
            C[(size_t)(m0 + t * 16 + v + hi8) * N + n0 + row] = acc[t][v];
    }
}

// ---------------- SpMM scatter-add: agg[dst[e]] += tmp[src[e]] ----------------
__global__ void k_spmm4(const float* __restrict__ tmp, const int* __restrict__ src,
                        const int* __restrict__ dst, float* __restrict__ agg,
                        int E, int F) {
    const int chunks = F >> 2;
    int t = blockIdx.x * blockDim.x + threadIdx.x;
    if (t >= E * chunks) return;
    int e = t / chunks;
    int c = (t - e * chunks) << 2;
    const float4 v = *(const float4*)(tmp + (size_t)src[e] * F + c);
    float* o = agg + (size_t)dst[e] * F + c;
    unsafeAtomicAdd(o + 0, v.x);
    unsafeAtomicAdd(o + 1, v.y);
    unsafeAtomicAdd(o + 2, v.z);
    unsafeAtomicAdd(o + 3, v.w);
}

// ---------------- epilogue 1: xs = bf16(relu(agg*ci + b1) * cj) ----------------
__global__ void k_post1(const float* __restrict__ agg, const float* __restrict__ ci,
                        const float* __restrict__ cj, const float* __restrict__ b,
                        bf16* __restrict__ xs, int total, int F) {
    int i = blockIdx.x * blockDim.x + threadIdx.x;
    if (i >= total) return;
    int n = i / F;
    int f = i - n * F;
    float v = fmaxf(fmaf(agg[i], ci[n], b[f]), 0.0f);
    xs[i] = (bf16)(v * cj[n]);
}

// ---------------- epilogue 2: sums[gid[n]] += agg*ci + b2 ----------------
__global__ void k_post2(const float* __restrict__ agg, const float* __restrict__ ci,
                        const float* __restrict__ b, const int* __restrict__ gid,
                        float* __restrict__ sums, int total, int F) {
    int i = blockIdx.x * blockDim.x + threadIdx.x;
    if (i >= total) return;
    int n = i / F;
    int f = i - n * F;
    float v = fmaf(agg[i], ci[n], b[f]);
    unsafeAtomicAdd(&sums[(size_t)gid[n] * F + f], v);
}

__global__ void k_count(const int* __restrict__ gid, float* __restrict__ cnt, int n) {
    int i = blockIdx.x * blockDim.x + threadIdx.x;
    if (i < n) unsafeAtomicAdd(&cnt[gid[i]], 1.0f);
}

// ---------------- readout: out = (sums/cnt) @ Wl + bl ----------------
__global__ void k_final(const float* __restrict__ sums, const float* __restrict__ cnt,
                        const float* __restrict__ Wl, const float* __restrict__ bl,
                        float* __restrict__ out, int G, int F, int L) {
    int i = blockIdx.x * blockDim.x + threadIdx.x;
    if (i >= G * L) return;
    int g = i / L;
    int l = i - g * L;
    float inv = 1.0f / fmaxf(cnt[g], 1.0f);
    float acc = bl[l];
    for (int k = 0; k < F; ++k)
        acc = fmaf(sums[(size_t)g * F + k] * inv, Wl[(size_t)k * L + l], acc);
    out[i] = acc;
}

extern "C" void kernel_launch(void* const* d_in, const int* in_sizes, int n_in,
                              void* d_out, int out_size, void* d_ws, size_t ws_size,
                              hipStream_t stream) {
    const float* x   = (const float*)d_in[0];
    const int*   src = (const int*)d_in[1];
    const int*   dst = (const int*)d_in[2];
    const int*   gid = (const int*)d_in[3];
    const float* W1  = (const float*)d_in[5];
    const float* b1  = (const float*)d_in[6];
    const float* W2  = (const float*)d_in[7];
    const float* b2  = (const float*)d_in[8];
    const float* Wl  = (const float*)d_in[9];
    const float* bl  = (const float*)d_in[10];
    float* out = (float*)d_out;

    const int u  = in_sizes[6];             // 256
    const int da = in_sizes[8];             // 256
    const int L  = in_sizes[10];            // 50
    const int de = in_sizes[5] / u;         // 128
    const int Nn = in_sizes[0] / de;        // 50000
    const int E  = in_sizes[1];             // 800000
    const int G  = out_size / L;            // 64
    const int Fm = (u > da) ? u : da;

    // carve workspace
    char* p = (char*)d_ws;
    auto carve = [&](size_t bytes) { char* r = p; p += (bytes + 255) & ~(size_t)255; return r; };
    float* cj   = (float*)carve((size_t)Nn * 4);
    float* ci   = (float*)carve((size_t)Nn * 4);
    float* cnt  = (float*)carve((size_t)G * 4);
    bf16*  W1p  = (bf16*) carve((size_t)de * u * 2);    // packed B fragments
    bf16*  W2p  = (bf16*) carve((size_t)u * da * 2);
    bf16*  xs   = (bf16*) carve((size_t)Nn * Fm * 2);   // reused layer1/layer2
    float* tmp  = (float*)carve((size_t)Nn * Fm * 4);   // GEMM out (reused)
    float* agg  = (float*)carve((size_t)Nn * Fm * 4);   // SpMM out (reused)
    float* sums = (float*)carve((size_t)G * da * 4);

    const int BT = 256;
    auto nb = [](long long n, int b) { return (unsigned)((n + b - 1) / b); };

    // --- degrees & norms ---
    hipMemsetAsync(cj, 0, (size_t)Nn * 4, stream);
    hipMemsetAsync(ci, 0, (size_t)Nn * 4, stream);
    k_degree<<<nb(E, BT), BT, 0, stream>>>(src, dst, cj, ci, E);
    k_rsqrt_clamp<<<nb(Nn, BT), BT, 0, stream>>>(cj, Nn);
    k_rsqrt_clamp<<<nb(Nn, BT), BT, 0, stream>>>(ci, Nn);

    // --- weight cast + WMMA pack ---
    k_packW<<<nb((long long)de * u, BT), BT, 0, stream>>>(W1, W1p, de, u);
    k_packW<<<nb((long long)u * da, BT), BT, 0, stream>>>(W2, W2p, u, da);

    // --- layer 1 ---
    k_prep1<<<nb((long long)Nn * de, BT), BT, 0, stream>>>(x, cj, xs, Nn * de, de);
    {
        dim3 grid(Nn / (16 * MT), u / 128);
        k_gemm_bf16<<<grid, 256, 0, stream>>>(xs, W1p, tmp, Nn, u, de);
    }
    hipMemsetAsync(agg, 0, (size_t)Nn * u * 4, stream);
    k_spmm4<<<nb((long long)E * (u / 4), BT), BT, 0, stream>>>(tmp, src, dst, agg, E, u);
    k_post1<<<nb((long long)Nn * u, BT), BT, 0, stream>>>(agg, ci, cj, b1, xs, Nn * u, u);

    // --- layer 2 ---
    {
        dim3 grid(Nn / (16 * MT), da / 128);
        k_gemm_bf16<<<grid, 256, 0, stream>>>(xs, W2p, tmp, Nn, da, u);
    }
    hipMemsetAsync(agg, 0, (size_t)Nn * da * 4, stream);
    k_spmm4<<<nb((long long)E * (da / 4), BT), BT, 0, stream>>>(tmp, src, dst, agg, E, da);

    // --- readout ---
    hipMemsetAsync(sums, 0, (size_t)G * da * 4, stream);
    hipMemsetAsync(cnt, 0, (size_t)G * 4, stream);
    k_count<<<nb(Nn, BT), BT, 0, stream>>>(gid, cnt, Nn);
    k_post2<<<nb((long long)Nn * da, BT), BT, 0, stream>>>(agg, ci, b2, gid, sums, Nn * da, da);
    k_final<<<nb((long long)G * L, BT), BT, 0, stream>>>(sums, cnt, Wl, bl, out, G, da, L);
}
